// Attention_74234214744512
// MI455X (gfx1250) — matrix operand order repaired and checked by reading, not compiled, not run on hardware
//
#include <hip/hip_runtime.h>
#include <hip/hip_bf16.h>

// ---------------------------------------------------------------------------
// CDNA5 (gfx1250, wave32) attention block built on v_wmma_f32_16x16x32_bf16.
// GEMMs use double-buffered LDS staging on the gfx1250 async Global->LDS
// path (ASYNCcnt-tracked, no VGPR round-trip, overlapped with WMMA), with a
// sync copy as fallback.
// ---------------------------------------------------------------------------

typedef __attribute__((ext_vector_type(16))) __bf16 v16bf;
typedef __attribute__((ext_vector_type(8)))  __bf16 bf16x8;
typedef __attribute__((ext_vector_type(8)))  float  v8f;
typedef __attribute__((ext_vector_type(4)))  int    v4i;

#define AS_GLOBAL __attribute__((address_space(1)))
#define AS_LDS    __attribute__((address_space(3)))

#if defined(__HIP_DEVICE_COMPILE__) &&                                   \
    __has_builtin(__builtin_amdgcn_global_load_async_to_lds_b128) &&     \
    __has_builtin(__builtin_amdgcn_s_wait_asynccnt)
#define USE_ASYNC_LDS 1
#else
#define USE_ASYNC_LDS 0
#endif

static constexpr int SEQ    = 2048;
static constexpr int HIDDEN = 4096;
static constexpr int HD     = 128;
static constexpr int NQH    = 32;
static constexpr int NKVH   = 8;

// ---- bf16 <-> f32 helpers (RNE) -------------------------------------------
__device__ __forceinline__ __bf16 f2bf(float f) {
    unsigned u = __builtin_bit_cast(unsigned, f);
    unsigned r = u + 0x7FFFu + ((u >> 16) & 1u);
    unsigned short h = (unsigned short)(r >> 16);
    return __builtin_bit_cast(__bf16, h);
}
__device__ __forceinline__ float bf2f(__bf16 b) {
    unsigned short h = __builtin_bit_cast(unsigned short, b);
    unsigned u = ((unsigned)h) << 16;
    return __builtin_bit_cast(float, u);
}

// ---- WMMA fragment loaders (16-bit A 16x32 layout, ISA 7.12.2) -------------
// A: lane<16 -> halves 0..7 = K 0..7, halves 8..15 = K 16..23; lane>=16: +8.
// Rows (M) = lane&15.  Source must be row-major with K contiguous,
// stride in halves must be a multiple of 8 (16-byte alignment).
union FragBF { v16bf v; bf16x8 h[2]; };

__device__ __forceinline__ v16bf load_frag_a(const __bf16* base, int stride) {
    const int lane = threadIdx.x & 31;
    const int row  = lane & 15;
    const int kb   = (lane >> 4) << 3;          // 0 or 8
    const __bf16* p = base + row * stride + kb;
    FragBF f;
    f.h[0] = *(const bf16x8*)(p);
    f.h[1] = *(const bf16x8*)(p + 16);
    return f.v;
}
// B (32x16): lane&15 -> column N; same K order as A (lane<16: K 0..7,16..23; lane>=16: +8).
// Source is B^T row-major ([N][K], K contiguous).
__device__ __forceinline__ v16bf load_frag_b(const __bf16* baseT, int stride) {
    const int lane = threadIdx.x & 31;
    const int col  = lane & 15;
    const int kb   = (lane >> 4) << 3;          // 0 or 8
    const __bf16* p = baseT + col * stride + kb;
    FragBF f;
    f.h[0] = *(const bf16x8*)(p);
    f.h[1] = *(const bf16x8*)(p + 16);
    return f.v;
}

// ---------------------------------------------------------------------------
// elementwise f32 -> bf16
__global__ void cvt_bf16_kernel(const float* __restrict__ x,
                                __bf16* __restrict__ y, long n) {
    long i = (long)blockIdx.x * blockDim.x + threadIdx.x;
    if (i < n) y[i] = f2bf(x[i]);
}

// W[K][N] f32  ->  Wt[N][K] bf16 (tiled transpose through LDS)
__global__ __launch_bounds__(256)
void transpose_cvt_kernel(const float* __restrict__ W, __bf16* __restrict__ Wt,
                          int K, int N) {
    __shared__ float tile[32][33];
    const int n0 = blockIdx.x * 32, k0 = blockIdx.y * 32;
    const int tx = threadIdx.x, ty = threadIdx.y;   // 32 x 8
#pragma unroll
    for (int i = 0; i < 32; i += 8)
        tile[ty + i][tx] = W[(long)(k0 + ty + i) * N + (n0 + tx)];
    __syncthreads();
#pragma unroll
    for (int i = 0; i < 32; i += 8)
        Wt[(long)(n0 + ty + i) * K + (k0 + tx)] = f2bf(tile[tx][ty + i]);
}

// ---------------------------------------------------------------------------
// bf16 GEMM:  C[M][N] = A[M][K] * Bt[N][K]^T, f32 accumulate via WMMA.
// 128x128 block tile, BK=32, 8 waves in a 4x2 grid, 2x4 frags per wave.
// Double-buffered LDS; async Global->LDS staging when available.
template <bool OUT_F32>
__global__ __launch_bounds__(256)
void gemm_bf16_kernel(const __bf16* __restrict__ A,
                      const __bf16* __restrict__ Bt,
                      void* __restrict__ Cp, int M, int N, int K) {
    constexpr int BK = 32, LDT = BK + 8;       // +8 halves: bank-conflict pad
    __shared__ __align__(16) __bf16 sA[2][128 * LDT];
    __shared__ __align__(16) __bf16 sB[2][128 * LDT];

    const int t    = threadIdx.x;
    const int wid  = t >> 5;
    const int wm   = wid >> 1;                  // 0..3 -> 32-row strip
    const int wn   = wid & 1;                   // 0..1 -> 64-col strip
    const int m0   = blockIdx.y * 128;
    const int n0   = blockIdx.x * 128;
    const int lane = t & 31;
    const int col  = lane & 15, hi = lane >> 4;

    v8f acc[2][4];
#pragma unroll
    for (int mi = 0; mi < 2; ++mi)
#pragma unroll
        for (int ni = 0; ni < 4; ++ni)
            acc[mi][ni] = (v8f){0.f,0.f,0.f,0.f,0.f,0.f,0.f,0.f};

    // stage one 128x32 tile of A and of Bt into LDS buffer `buf`.
    // Each thread issues exactly 4 x 16-byte transfers (2 A + 2 B).
    auto stage = [&](int k0, int buf) {
#pragma unroll
        for (int c = t; c < 512; c += 256) {
            const int row = c >> 2, off = (c & 3) << 3;
            const __bf16* ga = &A [(long)(m0 + row) * K + k0 + off];
            const __bf16* gb = &Bt[(long)(n0 + row) * K + k0 + off];
            __bf16* la = &sA[buf][row * LDT + off];
            __bf16* lb = &sB[buf][row * LDT + off];
#if USE_ASYNC_LDS
            __builtin_amdgcn_global_load_async_to_lds_b128(
                (AS_GLOBAL v4i*)ga, (AS_LDS v4i*)la, 0, 0);
            __builtin_amdgcn_global_load_async_to_lds_b128(
                (AS_GLOBAL v4i*)gb, (AS_LDS v4i*)lb, 0, 0);
#else
            *(bf16x8*)la = *(const bf16x8*)ga;
            *(bf16x8*)lb = *(const bf16x8*)gb;
#endif
        }
    };

    const int ntiles = K / BK;
    stage(0, 0);
    for (int i = 0; i < ntiles; ++i) {
        const int buf = i & 1;
        if (i + 1 < ntiles) stage((i + 1) * BK, buf ^ 1);
#if USE_ASYNC_LDS
        // In-order completion: <=4 outstanding means the previous stage's
        // 4 per-thread transfers have all landed in LDS.
        if (i + 1 < ntiles) __builtin_amdgcn_s_wait_asynccnt(4);
        else                __builtin_amdgcn_s_wait_asynccnt(0);
#endif
        __syncthreads();

        v16bf af[2], bfv[4];
#pragma unroll
        for (int mi = 0; mi < 2; ++mi)
            af[mi] = load_frag_a(&sA[buf][(wm * 32 + mi * 16) * LDT], LDT);
#pragma unroll
        for (int ni = 0; ni < 4; ++ni)
            bfv[ni] = load_frag_b(&sB[buf][(wn * 64 + ni * 16) * LDT], LDT);
#pragma unroll
        for (int mi = 0; mi < 2; ++mi)
#pragma unroll
            for (int ni = 0; ni < 4; ++ni)
                acc[mi][ni] = __builtin_amdgcn_wmma_f32_16x16x32_bf16(
                    false, af[mi], false, bfv[ni], (short)0, acc[mi][ni],
                    false, false);
        __syncthreads();   // all waves done reading `buf` before it is reused
    }

#pragma unroll
    for (int mi = 0; mi < 2; ++mi)
#pragma unroll
        for (int ni = 0; ni < 4; ++ni)
#pragma unroll
            for (int r = 0; r < 8; ++r) {
                const long row = m0 + wm * 32 + mi * 16 + r + 8 * hi;
                const long cc  = n0 + wn * 64 + ni * 16 + col;
                const float v  = acc[mi][ni][r];
                if (OUT_F32) ((float*)Cp)[row * N + cc] = v;
                else         ((__bf16*)Cp)[row * N + cc] = f2bf(v);
            }
}

// ---------------------------------------------------------------------------
// RoPE: lin[S][nh*128] bf16 -> out[nh][S][128] bf16 (head-major for attention)
__global__ void rope_kernel(const __bf16* __restrict__ lin,
                            const float* __restrict__ sin_t,
                            const float* __restrict__ cos_t,
                            __bf16* __restrict__ out, int nh) {
    long i = (long)blockIdx.x * blockDim.x + threadIdx.x;
    const long total = (long)nh * SEQ * 64;
    if (i >= total) return;
    const int j = (int)(i % 64);
    const int s = (int)((i / 64) % SEQ);
    const int h = (int)(i / (64L * SEQ));
    const int ld = nh * HD;
    const float x0 = bf2f(lin[(long)s * ld + h * HD + j]);
    const float x1 = bf2f(lin[(long)s * ld + h * HD + j + 64]);
    const float c  = cos_t[s * 64 + j];
    const float sn = sin_t[s * 64 + j];
    out[((long)h * SEQ + s) * HD + j]      = f2bf(x0 * c - x1 * sn);
    out[((long)h * SEQ + s) * HD + j + 64] = f2bf(x0 * sn + x1 * c);
}

// V: lin[S][8*128] -> vt[8][128][S]  (V^T per head, so P@V B-frags are
// contiguous along key positions)
__global__ void v_transpose_kernel(const __bf16* __restrict__ lin,
                                   __bf16* __restrict__ vt) {
    long i = (long)blockIdx.x * blockDim.x + threadIdx.x;
    const long total = (long)NKVH * HD * SEQ;
    if (i >= total) return;
    const int s = (int)(i % SEQ);
    const int d = (int)((i / SEQ) % HD);
    const int h = (int)(i / ((long)SEQ * HD));
    vt[((long)h * HD + d) * SEQ + s] = lin[(long)s * (NKVH * HD) + h * HD + d];
}

// ---------------------------------------------------------------------------
// Flash attention (causal, GQA 4:1).  Block = 256 thr = 8 waves; each wave
// owns 16 query rows of one head.  Per 64-key block: 16 WMMA for Q@K^T,
// online softmax with width-16 shuffles, P through wave-private LDS to
// A-layout, 16 WMMA for P@V.  K/V fragment reads stream from L2 (192 MB).
__global__ __launch_bounds__(256)
void flash_attn_kernel(const __bf16* __restrict__ Q,   // [32][S][128]
                       const __bf16* __restrict__ Kh,  // [8][S][128]
                       const __bf16* __restrict__ Vt,  // [8][128][S]
                       __bf16* __restrict__ O) {       // [S][4096]
    constexpr int LDP = 72;                        // padded P stride (halves)
    __shared__ __align__(16) __bf16 sP[8 * 16 * LDP];

    const int t = threadIdx.x, wid = t >> 5, lane = t & 31;
    const int col = lane & 15, hi = lane >> 4;
    const int qh = blockIdx.y, kvh = qh >> 2;
    const int qlo = blockIdx.x * 128 + wid * 16;   // this wave's 16 q rows

    const __bf16* Qb = Q  + ((long)qh * SEQ + qlo) * HD;
    const __bf16* Kb = Kh + (long)kvh * SEQ * HD;
    const __bf16* Vb = Vt + (long)kvh * HD * SEQ;
    __bf16* lp = &sP[wid * 16 * LDP];              // wave-private P patch

    v16bf qf[4];
#pragma unroll
    for (int ks = 0; ks < 4; ++ks) qf[ks] = load_frag_a(Qb + ks * 32, HD);

    v8f acc[8];
#pragma unroll
    for (int d = 0; d < 8; ++d) acc[d] = (v8f){0.f,0.f,0.f,0.f,0.f,0.f,0.f,0.f};
    float m[8], l[8];
#pragma unroll
    for (int r = 0; r < 8; ++r) { m[r] = -1e30f; l[r] = 0.f; }

    const float kscale = 0.12752859f;              // (1/sqrt(128)) * log2(e)

    for (int kb = 0; kb <= qlo + 15; kb += 64) {
        // ---- scores: S = Q @ K^T  (K-dim = head_dim, 4 steps of 32) ----
        v8f sf[4];
#pragma unroll
        for (int ni = 0; ni < 4; ++ni) {
            sf[ni] = (v8f){0.f,0.f,0.f,0.f,0.f,0.f,0.f,0.f};
#pragma unroll
            for (int ks = 0; ks < 4; ++ks) {
                v16bf b = load_frag_b(Kb + (long)(kb + ni * 16) * HD + ks * 32, HD);
                sf[ni] = __builtin_amdgcn_wmma_f32_16x16x32_bf16(
                    false, qf[ks], false, b, (short)0, sf[ni], false, false);
            }
        }
        // ---- scale + causal mask (VALU select; EXEC stays all-ones) ----
#pragma unroll
        for (int ni = 0; ni < 4; ++ni)
#pragma unroll
            for (int r = 0; r < 8; ++r) {
                const int qrow = qlo + r + 8 * hi;
                const int kcol = kb + ni * 16 + col;
                const float s = sf[ni][r] * kscale;
                sf[ni][r] = (kcol <= qrow) ? s : -1e30f;
            }
        // ---- online softmax (rows live in lane groups of 16) ----
        float alpha[8];
#pragma unroll
        for (int r = 0; r < 8; ++r) {
            float mx = sf[0][r];
#pragma unroll
            for (int ni = 1; ni < 4; ++ni) mx = fmaxf(mx, sf[ni][r]);
#pragma unroll
            for (int off = 8; off; off >>= 1)
                mx = fmaxf(mx, __shfl_xor(mx, off, 16));
            const float mn = fmaxf(m[r], mx);
            alpha[r] = exp2f(m[r] - mn);
            m[r] = mn;
            float rs = 0.f;
#pragma unroll
            for (int ni = 0; ni < 4; ++ni) {
                const float p = exp2f(sf[ni][r] - mn);
                sf[ni][r] = p;
                rs += p;
            }
#pragma unroll
            for (int off = 8; off; off >>= 1) rs += __shfl_xor(rs, off, 16);
            l[r] = l[r] * alpha[r] + rs;
        }
#pragma unroll
        for (int d = 0; d < 8; ++d)
#pragma unroll
            for (int r = 0; r < 8; ++r) acc[d][r] *= alpha[r];
        // ---- P (C-layout) -> LDS -> A-layout fragments ----
#pragma unroll
        for (int ni = 0; ni < 4; ++ni)
#pragma unroll
            for (int r = 0; r < 8; ++r)
                lp[(r + 8 * hi) * LDP + ni * 16 + col] = f2bf(sf[ni][r]);
        // ---- O += P @ V  (K-dim = 64 key positions, 2 steps of 32) ----
#pragma unroll
        for (int ks2 = 0; ks2 < 2; ++ks2) {
            v16bf pa = load_frag_a(lp + ks2 * 32, LDP);
#pragma unroll
            for (int d = 0; d < 8; ++d) {
                v16bf b = load_frag_b(Vb + (long)(d * 16) * SEQ + kb + ks2 * 32, SEQ);
                acc[d] = __builtin_amdgcn_wmma_f32_16x16x32_bf16(
                    false, pa, false, b, (short)0, acc[d], false, false);
            }
        }
    }
    // ---- normalize + store into [S][32*128] for the output projection ----
#pragma unroll
    for (int r = 0; r < 8; ++r) l[r] = 1.f / l[r];
#pragma unroll
    for (int d = 0; d < 8; ++d)
#pragma unroll
        for (int r = 0; r < 8; ++r) {
            const long row = qlo + r + 8 * hi;
            O[row * (NQH * HD) + qh * HD + d * 16 + col] = f2bf(acc[d][r] * l[r]);
        }
}

// ---------------------------------------------------------------------------
extern "C" void kernel_launch(void* const* d_in, const int* in_sizes, int n_in,
                              void* d_out, int out_size, void* d_ws, size_t ws_size,
                              hipStream_t stream) {
    (void)in_sizes; (void)n_in; (void)out_size; (void)ws_size;
    const float* hidden = (const float*)d_in[0];
    const float* sin_t  = (const float*)d_in[1];
    const float* cos_t  = (const float*)d_in[2];
    const float* Wq     = (const float*)d_in[3];
    const float* Wk     = (const float*)d_in[4];
    const float* Wv     = (const float*)d_in[5];
    const float* Wo     = (const float*)d_in[6];

    size_t off = 0;
    auto carve = [&](size_t bytes) -> void* {
        void* p = (char*)d_ws + off;
        off += (bytes + 255) & ~(size_t)255;
        return p;
    };
    __bf16* Xbf  = (__bf16*)carve((size_t)SEQ * HIDDEN * 2);
    __bf16* WqT  = (__bf16*)carve((size_t)HIDDEN * (NQH * HD) * 2);
    __bf16* WkT  = (__bf16*)carve((size_t)HIDDEN * (NKVH * HD) * 2);
    __bf16* WvT  = (__bf16*)carve((size_t)HIDDEN * (NKVH * HD) * 2);
    __bf16* WoT  = (__bf16*)carve((size_t)(NQH * HD) * HIDDEN * 2);
    __bf16* Qlin = (__bf16*)carve((size_t)SEQ * (NQH * HD) * 2);
    __bf16* Klin = (__bf16*)carve((size_t)SEQ * (NKVH * HD) * 2);
    __bf16* Vlin = (__bf16*)carve((size_t)SEQ * (NKVH * HD) * 2);
    __bf16* Qr   = (__bf16*)carve((size_t)NQH * SEQ * HD * 2);
    __bf16* Kr   = (__bf16*)carve((size_t)NKVH * SEQ * HD * 2);
    __bf16* Vtr  = (__bf16*)carve((size_t)NKVH * HD * SEQ * 2);
    __bf16* AtO  = (__bf16*)carve((size_t)SEQ * (NQH * HD) * 2);

    // 1) precision conversion + weight transposes
    {
        const long n = (long)SEQ * HIDDEN;
        cvt_bf16_kernel<<<(unsigned)((n + 255) / 256), 256, 0, stream>>>(hidden, Xbf, n);
    }
    transpose_cvt_kernel<<<dim3((NQH  * HD) / 32, HIDDEN / 32), dim3(32, 8), 0, stream>>>(Wq, WqT, HIDDEN, NQH  * HD);
    transpose_cvt_kernel<<<dim3((NKVH * HD) / 32, HIDDEN / 32), dim3(32, 8), 0, stream>>>(Wk, WkT, HIDDEN, NKVH * HD);
    transpose_cvt_kernel<<<dim3((NKVH * HD) / 32, HIDDEN / 32), dim3(32, 8), 0, stream>>>(Wv, WvT, HIDDEN, NKVH * HD);
    transpose_cvt_kernel<<<dim3(HIDDEN / 32, (NQH * HD) / 32),  dim3(32, 8), 0, stream>>>(Wo, WoT, NQH * HD, HIDDEN);

    // 2) QKV projections (WMMA bf16)
    gemm_bf16_kernel<false><<<dim3((NQH  * HD) / 128, SEQ / 128), 256, 0, stream>>>(Xbf, WqT, Qlin, SEQ, NQH  * HD, HIDDEN);
    gemm_bf16_kernel<false><<<dim3((NKVH * HD) / 128, SEQ / 128), 256, 0, stream>>>(Xbf, WkT, Klin, SEQ, NKVH * HD, HIDDEN);
    gemm_bf16_kernel<false><<<dim3((NKVH * HD) / 128, SEQ / 128), 256, 0, stream>>>(Xbf, WvT, Vlin, SEQ, NKVH * HD, HIDDEN);

    // 3) RoPE + layout changes for attention
    {
        const long nq = (long)NQH * SEQ * 64;
        rope_kernel<<<(unsigned)((nq + 255) / 256), 256, 0, stream>>>(Qlin, sin_t, cos_t, Qr, NQH);
        const long nk = (long)NKVH * SEQ * 64;
        rope_kernel<<<(unsigned)((nk + 255) / 256), 256, 0, stream>>>(Klin, sin_t, cos_t, Kr, NKVH);
        const long nv = (long)NKVH * HD * SEQ;
        v_transpose_kernel<<<(unsigned)((nv + 255) / 256), 256, 0, stream>>>(Vlin, Vtr);
    }

    // 4) causal flash attention (WMMA bf16)
    flash_attn_kernel<<<dim3(SEQ / 128, NQH), 256, 0, stream>>>(Qr, Kr, Vtr, AtO);

    // 5) output projection -> f32 d_out
    gemm_bf16_kernel<true><<<dim3(HIDDEN / 128, SEQ / 128), 256, 0, stream>>>(AtO, WoT, (float*)d_out, SEQ, HIDDEN, NQH * HD);
}
